// multi_head_attention_44298292691497
// MI455X (gfx1250) — matrix operand order repaired and checked by reading, not compiled, not run on hardware
//
#include <hip/hip_runtime.h>
#include <hip/hip_bf16.h>
#include <cstdint>
#include <cstddef>

// ---------------------------------------------------------------------------
// MI455X / gfx1250 multi-head attention (column-softmax variant), f16 WMMA.
// ---------------------------------------------------------------------------

typedef _Float16 h16_t __attribute__((ext_vector_type(16)));
typedef _Float16 h8_t  __attribute__((ext_vector_type(8)));
typedef _Float16 h4_t  __attribute__((ext_vector_type(4)));
typedef float    f8_t  __attribute__((ext_vector_type(8)));

static __device__ __forceinline__ f8_t wmma_f16(h16_t a, h16_t b, f8_t c) {
  // D = A(16x32 f16) * B(32x16 f16) + C(16x16 f32)
  return __builtin_amdgcn_wmma_f32_16x16x32_f16(false, a, false, b,
                                                (short)0, c, false, false);
}

static __device__ __forceinline__ h16_t pack16(h8_t lo, h8_t hi) {
  h16_t r;
#pragma unroll
  for (int e = 0; e < 8; ++e) { r[e] = lo[e]; r[8 + e] = hi[e]; }
  return r;
}

static __device__ __forceinline__ h8_t cvt8(float4 a, float4 b) {
  h8_t o;
  o[0] = (_Float16)a.x; o[1] = (_Float16)a.y;
  o[2] = (_Float16)a.z; o[3] = (_Float16)a.w;
  o[4] = (_Float16)b.x; o[5] = (_Float16)b.y;
  o[6] = (_Float16)b.z; o[7] = (_Float16)b.w;
  return o;
}

// Problem constants
#define BB 2
#define TT 2048
#define DD 1024
#define HH 16
#define ND 64
#define NTOK (BB * TT)          // 4096
#define NELEM ((size_t)NTOK * DD)

// ---------------------------------------------------------------------------
// fp32 -> fp16 elementwise convert (vectorized float4 -> h4)
// ---------------------------------------------------------------------------
__global__ void cvt_f32_f16(const float* __restrict__ src,
                            _Float16* __restrict__ dst, int n4) {
  int i = blockIdx.x * blockDim.x + threadIdx.x;
  if (i < n4) {
    float4 v = reinterpret_cast<const float4*>(src)[i];
    h4_t o;
    o[0] = (_Float16)v.x; o[1] = (_Float16)v.y;
    o[2] = (_Float16)v.z; o[3] = (_Float16)v.w;
    reinterpret_cast<h4_t*>(dst)[i] = o;
  }
}

// ---------------------------------------------------------------------------
// GEMM: Y = A(f16)[4096x1024] @ W(f32)[1024x1024]^T + bias
//  OUT_MODE 0: write f16, head-split layout [B][H][T][nd]
//  OUT_MODE 1: write f32, flat [4096][1024]   (final projection -> d_out)
// Block 128x128x32; 8 waves, each wave a 64(M)x32(N) macro-tile (4x2 WMMA).
// Double-buffered LDS staging, one barrier per K-step, prefetch 2 ahead.
// ---------------------------------------------------------------------------
template <int OUT_MODE>
__global__ __launch_bounds__(256) void gemm_wmma(
    const _Float16* __restrict__ A, const float* __restrict__ W,
    const float* __restrict__ bias, _Float16* __restrict__ outH,
    float* __restrict__ outF) {
  __shared__ _Float16 As[2][128][40];  // +8 pad: 80B rows keep 16B alignment
  __shared__ _Float16 Ws[2][128][40];

  const int tid  = threadIdx.x;
  const int wid  = tid >> 5, lane = tid & 31;
  const int lh   = lane >> 4, ll = lane & 15;
  const int wm   = wid & 1,  wn = wid >> 1;       // 2 x 4 wave grid
  const int m0   = blockIdx.y * 128;
  const int n0   = blockIdx.x * 128;
  const int lr   = tid >> 1;                      // 0..127 (tile row)
  const int lc   = tid & 1;                       // 0..1   (16-col half)

  const _Float16* aSrc = A + (size_t)(m0 + lr) * DD + lc * 16;
  const float*    wSrc = W + (size_t)(n0 + lr) * DD + lc * 16;

  // Preload K-tile 0 into buffer 0
  {
    const h8_t* as = reinterpret_cast<const h8_t*>(aSrc);
    const float4* wsrc = reinterpret_cast<const float4*>(wSrc);
    h8_t a0 = as[0], a1 = as[1];
    float4 w0 = wsrc[0], w1 = wsrc[1], w2 = wsrc[2], w3 = wsrc[3];
    h8_t* ad = reinterpret_cast<h8_t*>(&As[0][lr][lc * 16]);
    h8_t* wd = reinterpret_cast<h8_t*>(&Ws[0][lr][lc * 16]);
    ad[0] = a0; ad[1] = a1;
    wd[0] = cvt8(w0, w1); wd[1] = cvt8(w2, w3);
  }
  __syncthreads();

  f8_t acc[4][2] = {};

  for (int kb = 0; kb < 32; ++kb) {
    const int cur = kb & 1;
    const int nxt = cur ^ 1;
    const bool have_next = (kb + 1) < 32;

    // Issue next tile's global loads first (overlap with WMMA below)
    h8_t na0, na1;
    float4 nw0, nw1, nw2, nw3;
    if (have_next) {
      const h8_t* as = reinterpret_cast<const h8_t*>(aSrc + (kb + 1) * 32);
      const float4* wsrc =
          reinterpret_cast<const float4*>(wSrc + (kb + 1) * 32);
      na0 = as[0]; na1 = as[1];
      nw0 = wsrc[0]; nw1 = wsrc[1]; nw2 = wsrc[2]; nw3 = wsrc[3];
    }
    if (kb + 2 < 32) {                       // -> global_prefetch_b8
      __builtin_prefetch(aSrc + (kb + 2) * 32, 0, 1);
      __builtin_prefetch(wSrc + (kb + 2) * 32, 0, 1);
    }

    // A fragments: lane = M row (ll), K half at lh*16 (16 contiguous f16)
    h16_t af[4];
#pragma unroll
    for (int i = 0; i < 4; ++i) {
      const int row = wm * 64 + i * 16 + ll;
      h8_t lo = *reinterpret_cast<const h8_t*>(&As[cur][row][lh * 16]);
      h8_t hi = *reinterpret_cast<const h8_t*>(&As[cur][row][lh * 16 + 8]);
      af[i] = pack16(lo, hi);
    }
    // B fragments: lane = N col (ll), K half at lh*16 (16 contiguous f16)
    h16_t bf[2];
#pragma unroll
    for (int j = 0; j < 2; ++j) {
      const int row = wn * 32 + j * 16 + ll;
      const h8_t* p = reinterpret_cast<const h8_t*>(&Ws[cur][row][lh * 16]);
      bf[j] = pack16(p[0], p[1]);
    }
#pragma unroll
    for (int i = 0; i < 4; ++i)
#pragma unroll
      for (int j = 0; j < 2; ++j)
        acc[i][j] = wmma_f16(af[i], bf[j], acc[i][j]);

    if (have_next) {
      h8_t* ad = reinterpret_cast<h8_t*>(&As[nxt][lr][lc * 16]);
      h8_t* wd = reinterpret_cast<h8_t*>(&Ws[nxt][lr][lc * 16]);
      ad[0] = na0; ad[1] = na1;
      wd[0] = cvt8(nw0, nw1); wd[1] = cvt8(nw2, nw3);
    }
    __syncthreads();
  }

  // Store: C layout -> VGPR r holds (M = r + 8*lh, N = ll)
  const int b_idx = m0 >> 11;               // block never crosses a batch
#pragma unroll
  for (int i = 0; i < 4; ++i) {
#pragma unroll
    for (int j = 0; j < 2; ++j) {
      const int gn = n0 + wn * 32 + j * 16 + ll;
      const float bj = bias[gn];
      const int t0 = (m0 & 2047) + wm * 64 + i * 16 + lh * 8;
      if (OUT_MODE == 0) {
        const int hd = gn >> 6, d = gn & 63;
        _Float16* p =
            outH + (((size_t)b_idx * HH + hd) * TT + t0) * ND + d;
#pragma unroll
        for (int r = 0; r < 8; ++r) p[(size_t)r * ND] =
            (_Float16)(acc[i][j][r] + bj);
      } else {
        float* p = outF + ((size_t)b_idx * TT + t0) * DD + gn;
#pragma unroll
        for (int r = 0; r < 8; ++r) p[(size_t)r * DD] = acc[i][j][r] + bj;
      }
    }
  }
}

// ---------------------------------------------------------------------------
// Pass 1: per-column (key axis) online softmax stats over the causal-masked
// score matrix. softmax is over the QUERY axis (axis=-2 in the reference).
// Grid: (B*H, 16 segments) x 256 threads; each wave owns one 16-col k-tile.
// ---------------------------------------------------------------------------
__global__ __launch_bounds__(256) void attn_stats(
    const _Float16* __restrict__ qh, const _Float16* __restrict__ kh,
    float* __restrict__ mstat, float* __restrict__ sstat) {
  const int bh   = blockIdx.x;
  const int seg  = blockIdx.y;
  const int tid  = threadIdx.x;
  const int wid  = tid >> 5, lane = tid & 31;
  const int lh   = lane >> 4, ll = lane & 15;
  const int kt   = seg * 8 + wid;       // 0..127
  const int k0   = kt * 16;
  const size_t base = (size_t)bh * TT * ND;
  const int kcol = k0 + ll;

  // K rows are the columns of z -> B operand; d = 64 in two K=32 steps.
  h16_t bf[2];
#pragma unroll
  for (int s = 0; s < 2; ++s) {
    const h8_t* p = reinterpret_cast<const h8_t*>(
        kh + base + (size_t)(k0 + ll) * ND + s * 32 + lh * 16);
    bf[s] = pack16(p[0], p[1]);
  }

  float m = -1e30f, ssum = 0.f;
  for (int q0 = k0; q0 < TT; q0 += 16) {
    const _Float16* qrow = qh + base + (size_t)(q0 + ll) * ND;
    if (q0 + 16 < TT)
      __builtin_prefetch(qrow + (size_t)16 * ND, 0, 1);
    h16_t af[2];
#pragma unroll
    for (int s = 0; s < 2; ++s) {
      h8_t lo = *reinterpret_cast<const h8_t*>(qrow + s * 32 + lh * 16);
      h8_t hi = *reinterpret_cast<const h8_t*>(qrow + s * 32 + 8 + lh * 16);
      af[s] = pack16(lo, hi);
    }
    f8_t c = {};
    c = wmma_f16(af[0], bf[0], c);
    c = wmma_f16(af[1], bf[1], c);

    float zv[8];
    float zmax = -1e30f;
#pragma unroll
    for (int r = 0; r < 8; ++r) {
      const int q = q0 + r + lh * 8;
      zv[r] = (q >= kcol) ? c[r] * 0.125f : -1e30f;   // 1/sqrt(64)
      zmax = fmaxf(zmax, zv[r]);
    }
    if (zmax > -1e29f) {
      const float nm = fmaxf(m, zmax);
      float a = 0.f;
#pragma unroll
      for (int r = 0; r < 8; ++r)
        a += (zv[r] > -1e29f) ? __expf(zv[r] - nm) : 0.f;
      ssum = ssum * __expf(m - nm) + a;
      m = nm;
    }
  }
  // Combine lane L with lane L+16 (same column, other 8 rows)
  const float m2 = __shfl_xor(m, 16);
  const float s2 = __shfl_xor(ssum, 16);
  const float nm = fmaxf(m, m2);
  const float S  = ssum * __expf(m - nm) + s2 * __expf(m2 - nm);
  if (lane < 16) {
    mstat[(size_t)bh * TT + kcol] = nm;
    sstat[(size_t)bh * TT + kcol] = S;
  }
}

// ---------------------------------------------------------------------------
// Pass 2: out[q,:] = sum_k exp(z[q,k]-m[k])/s[k] * V[k,:]
// Grid: (B*H, 16 q-blocks of 128) x 256; wave owns one 16-row q-tile.
// V is staged transposed in LDS; P goes through LDS (C->A layout fixup).
// ---------------------------------------------------------------------------
__global__ __launch_bounds__(256) void attn_out(
    const _Float16* __restrict__ qh, const _Float16* __restrict__ kh,
    const _Float16* __restrict__ vh, const float* __restrict__ mstat,
    const float* __restrict__ sstat, _Float16* __restrict__ ctx) {
  __shared__ _Float16 Vt[64][40];       // V chunk, transposed [d][k]
  __shared__ _Float16 Pb[8][16][40];    // per-wave probability tiles [q][k]

  const int bh  = blockIdx.x;
  const int qb  = blockIdx.y;
  const int tid = threadIdx.x;
  const int wid = tid >> 5, lane = tid & 31;
  const int lh  = lane >> 4, ll = lane & 15;
  const int q0  = qb * 128 + wid * 16;
  const size_t base = (size_t)bh * TT * ND;
  const int b = bh >> 4, hd = bh & 15;

  // Q fragments hoisted (d = 0..63 in two K=32 steps)
  h16_t qf[2];
#pragma unroll
  for (int s = 0; s < 2; ++s) {
    const _Float16* row = qh + base + (size_t)(q0 + ll) * ND + s * 32;
    h8_t lo = *reinterpret_cast<const h8_t*>(row + lh * 16);
    h8_t hi = *reinterpret_cast<const h8_t*>(row + 8 + lh * 16);
    qf[s] = pack16(lo, hi);
  }

  f8_t acc[4] = {};
  const int nchunk = qb * 4 + 4;        // k-chunks of 32 up to block max q

  for (int ch = 0; ch < nchunk; ++ch) {
    const int kb = ch * 32;
    // Stage V[kb..kb+32)[0..64) transposed into LDS
    {
      const int kr = tid >> 3;          // 0..31
      const int d0 = (tid & 7) * 8;
      h8_t v = *reinterpret_cast<const h8_t*>(
          vh + base + (size_t)(kb + kr) * ND + d0);
#pragma unroll
      for (int e = 0; e < 8; ++e) Vt[d0 + e][kr] = v[e];
    }
    __syncthreads();

    if (kb < q0 + 16) {                 // wave-uniform: EXEC stays all-1s
#pragma unroll
      for (int half = 0; half < 2; ++half) {
        const int k0 = kb + half * 16;
        if (k0 < q0 + 16) {
          h16_t bf2[2];
#pragma unroll
          for (int s = 0; s < 2; ++s) {
            const h8_t* p = reinterpret_cast<const h8_t*>(
                kh + base + (size_t)(k0 + ll) * ND + s * 32 + lh * 16);
            bf2[s] = pack16(p[0], p[1]);
          }
          f8_t c = {};
          c = wmma_f16(qf[0], bf2[0], c);
          c = wmma_f16(qf[1], bf2[1], c);
          const int kcol = k0 + ll;
          const float mk = mstat[(size_t)bh * TT + kcol];
          const float rk = 1.0f / sstat[(size_t)bh * TT + kcol];
#pragma unroll
          for (int r = 0; r < 8; ++r) {
            const int q = q0 + r + lh * 8;
            const float p =
                (q >= kcol) ? __expf(c[r] * 0.125f - mk) * rk : 0.f;
            Pb[wid][r + lh * 8][half * 16 + ll] = (_Float16)p;
          }
        } else {
#pragma unroll
          for (int r = 0; r < 8; ++r)
            Pb[wid][r + lh * 8][half * 16 + ll] = (_Float16)0.f;
        }
      }
      // P as A operand (same-wave LDS RAW: DS ops are in-order per wave)
      h8_t plo = *reinterpret_cast<const h8_t*>(&Pb[wid][ll][lh * 16]);
      h8_t phi = *reinterpret_cast<const h8_t*>(&Pb[wid][ll][lh * 16 + 8]);
      h16_t pf = pack16(plo, phi);
#pragma unroll
      for (int dt = 0; dt < 4; ++dt) {
        const h8_t* vp =
            reinterpret_cast<const h8_t*>(&Vt[dt * 16 + ll][lh * 16]);
        h16_t vf = pack16(vp[0], vp[1]);
        acc[dt] = wmma_f16(pf, vf, acc[dt]);
      }
    }
    __syncthreads();
  }

  // Store context f16 in flat [B][T][D] layout (col = h*64 + d)
#pragma unroll
  for (int dt = 0; dt < 4; ++dt) {
    const int col = hd * ND + dt * 16 + ll;
    _Float16* p = ctx + ((size_t)b * TT + q0 + lh * 8) * DD + col;
#pragma unroll
    for (int r = 0; r < 8; ++r) p[(size_t)r * DD] = (_Float16)acc[dt][r];
  }
}

// ---------------------------------------------------------------------------
extern "C" void kernel_launch(void* const* d_in, const int* in_sizes, int n_in,
                              void* d_out, int out_size, void* d_ws,
                              size_t ws_size, hipStream_t stream) {
  (void)in_sizes; (void)n_in; (void)out_size; (void)ws_size;
  const float* q  = (const float*)d_in[0];
  const float* k  = (const float*)d_in[1];
  const float* v  = (const float*)d_in[2];
  const float* Wq = (const float*)d_in[3];
  const float* bq = (const float*)d_in[4];
  const float* Wk = (const float*)d_in[5];
  const float* bk = (const float*)d_in[6];
  const float* Wv = (const float*)d_in[7];
  const float* bv = (const float*)d_in[8];
  const float* Wc = (const float*)d_in[9];
  const float* bc = (const float*)d_in[10];
  float* out = (float*)d_out;

  char* ws = (char*)d_ws;
  const size_t SZ16 = NELEM * sizeof(_Float16);      // 8 MB each
  _Float16* x16q = (_Float16*)(ws + 0 * SZ16);
  _Float16* x16k = (_Float16*)(ws + 1 * SZ16);
  _Float16* x16v = (_Float16*)(ws + 2 * SZ16);
  _Float16* qhh  = (_Float16*)(ws + 3 * SZ16);
  _Float16* khh  = (_Float16*)(ws + 4 * SZ16);
  _Float16* vhh  = (_Float16*)(ws + 5 * SZ16);
  _Float16* ctx  = (_Float16*)(ws + 6 * SZ16);
  float* mstat   = (float*)(ws + 7 * SZ16);
  float* sstat   = (float*)(ws + 7 * SZ16 + (size_t)BB * HH * TT * 4);

  const int n4 = (int)(NELEM / 4);
  const int cgrid = (n4 + 255) / 256;
  cvt_f32_f16<<<cgrid, 256, 0, stream>>>(q, x16q, n4);
  cvt_f32_f16<<<cgrid, 256, 0, stream>>>(k, x16k, n4);
  cvt_f32_f16<<<cgrid, 256, 0, stream>>>(v, x16v, n4);

  dim3 ggrid(DD / 128, NTOK / 128);  // (8, 32)
  gemm_wmma<0><<<ggrid, 256, 0, stream>>>(x16q, Wq, bq, qhh, nullptr);
  gemm_wmma<0><<<ggrid, 256, 0, stream>>>(x16k, Wk, bk, khh, nullptr);
  gemm_wmma<0><<<ggrid, 256, 0, stream>>>(x16v, Wv, bv, vhh, nullptr);

  attn_stats<<<dim3(BB * HH, 16), 256, 0, stream>>>(qhh, khh, mstat, sstat);
  attn_out<<<dim3(BB * HH, 16), 256, 0, stream>>>(qhh, khh, vhh, mstat, sstat,
                                                  ctx);

  gemm_wmma<1><<<ggrid, 256, 0, stream>>>(ctx, Wc, bc, nullptr, out);
}
